// FastAssociativeAttention_23278722744906
// MI455X (gfx1250) — matrix operand order repaired and checked by reading, not compiled, not run on hardware
//
#include <hip/hip_runtime.h>
#include <hip/hip_bf16.h>

// Problem constants (match reference)
#define Bb   2
#define Ss   2048
#define HID  2048
#define NH   16
#define HD   128
#define MTOK (Bb * Ss)                 // 4096
#define SCALE 0.08838834764831845f    // 1/sqrt(128)

typedef __attribute__((ext_vector_type(16))) __bf16 bf16x16;
typedef __attribute__((ext_vector_type(8)))  float  f32x8;

// ---------------------------------------------------------------------------
// CDNA5 async global->LDS copy (ASYNCcnt) with graceful fallbacks.
// ---------------------------------------------------------------------------
#if __has_builtin(__builtin_amdgcn_global_load_async_to_lds_b128)
#define ASYNC_MODE 2   // clang builtin
#elif defined(__gfx1250__)
#define ASYNC_MODE 1   // inline asm
#else
#define ASYNC_MODE 0   // synchronous copy
#endif

typedef int i32x4_vs __attribute__((vector_size(16)));
typedef __attribute__((address_space(1))) void* as1_void_p;
typedef __attribute__((address_space(3))) void* as3_void_p;
typedef __attribute__((address_space(1))) i32x4_vs* as1_v4i_p;
typedef __attribute__((address_space(3))) i32x4_vs* as3_v4i_p;

__device__ __forceinline__ void cp16(__bf16* l, const __bf16* g) {
#if ASYNC_MODE == 2
  as1_v4i_p gp = (as1_v4i_p)(as1_void_p)(void*)g;   // generic -> AS1 -> int4*
  as3_v4i_p lp = (as3_v4i_p)(as3_void_p)(void*)l;   // generic -> AS3 -> int4*
  __builtin_amdgcn_global_load_async_to_lds_b128(gp, lp, 0, 0);
#elif ASYNC_MODE == 1
  unsigned int loff = (unsigned int)(size_t)l;   // LDS aperture: low 32 bits
  asm volatile("global_load_async_to_lds_b128 %0, %1, off"
               :: "v"(loff), "v"(g) : "memory");
#else
  *(uint4*)l = *(const uint4*)g;
#endif
}

__device__ __forceinline__ void wait_async0() {
#if ASYNC_MODE == 2
#if __has_builtin(__builtin_amdgcn_s_wait_asynccnt)
  __builtin_amdgcn_s_wait_asynccnt(0);
#else
  asm volatile("s_wait_asynccnt 0x0" ::: "memory");
#endif
#elif ASYNC_MODE == 1
  asm volatile("s_wait_asynccnt 0x0" ::: "memory");
#endif
}

// ---------------------------------------------------------------------------
__device__ __forceinline__ f32x8 zero8() {
  f32x8 z;
#pragma unroll
  for (int i = 0; i < 8; ++i) z[i] = 0.0f;
  return z;
}

__device__ __forceinline__ unsigned int pack2bf16(float a, float b) {
  union { float f; unsigned int u; } ua, ub;
  ua.f = a; ub.f = b;
  unsigned int x = ua.u, y = ub.u;
  x = (x + 0x7FFFu + ((x >> 16) & 1u)) >> 16;       // RNE
  y = (y + 0x7FFFu + ((y >> 16) & 1u)) >> 16;
  return x | (y << 16);
}

// A-matrix 16x32 bf16 fragment: lanes 0-15 K=0..15,
// lanes 16-31 K=16..31; row = lane&15.
__device__ __forceinline__ bf16x16 load_a_frag(const __bf16* t, int ld) {
  const int lane = threadIdx.x & 31;
  const int row  = lane & 15;
  const int kb   = (lane >> 4) << 4;      // 0 or 16
  const __bf16* p = t + row * ld;
  bf16x16 f;
#pragma unroll
  for (int j = 0; j < 8; ++j) f[j]     = p[kb + j];
#pragma unroll
  for (int j = 0; j < 8; ++j) f[8 + j] = p[kb + 8 + j];
  return f;
}

// B-matrix 32x16 bf16 fragment, LDS-resident as [N=16][K=32]:
// lane holds column n=lane&15; lanes 0-15 K=0..15, lanes 16-31 K=16..31.
__device__ __forceinline__ bf16x16 load_b_frag(const __bf16* t, int ld) {
  const int lane = threadIdx.x & 31;
  const int n    = lane & 15;
  const int kb   = (lane >> 4) << 4;      // 0 or 16
  const __bf16* p = t + n * ld + kb;
  bf16x16 f;
#pragma unroll
  for (int j = 0; j < 16; ++j) f[j] = p[j];
  return f;
}

__device__ __forceinline__ f32x8 wmma_bf16(bf16x16 a, bf16x16 b, f32x8 c) {
  return __builtin_amdgcn_wmma_f32_16x16x32_bf16(false, a, false, b,
                                                 (short)0, c, false, false);
}

// ---------------------------------------------------------------------------
// Streaming fp32 -> bf16 (RNE) conversion, float4-vectorized.
// ---------------------------------------------------------------------------
__global__ __launch_bounds__(256) void f32_to_bf16(
    const float* __restrict__ in, __bf16* __restrict__ out, int n4)
{
  const int t = blockIdx.x * 256 + threadIdx.x;
  if (t < n4) {
    float4 v = ((const float4*)in)[t];
    uint2 p;
    p.x = pack2bf16(v.x, v.y);
    p.y = pack2bf16(v.z, v.w);
    ((uint2*)out)[t] = p;
  }
}

// ---------------------------------------------------------------------------
// C[M,N] = A[M,K] @ W[N,K]^T   (bf16 in, fp32 out), tile 128x128, K-step 32,
// double-buffered async global->LDS, 8 waves x 8 WMMA per K-step.
// ---------------------------------------------------------------------------
__global__ __launch_bounds__(256) void gemm_bf16nt(
    const __bf16* __restrict__ A, const __bf16* __restrict__ W,
    float* __restrict__ C, int M, int N, int K)
{
  __shared__ __bf16 As[2][128 * 32];
  __shared__ __bf16 Bs[2][128 * 32];

  const int tid  = threadIdx.x;
  const int m0   = blockIdx.y * 128;
  const int n0   = blockIdx.x * 128;
  const int lrow = tid >> 1;
  const int lcol = (tid & 1) * 16;
  const int w    = tid >> 5;
  const int wm   = (w >> 2) * 64;
  const int wn   = (w & 3) * 32;
  const int lane = tid & 31;

  const __bf16* ga = A + (size_t)(m0 + lrow) * K + lcol;
  const __bf16* gw = W + (size_t)(n0 + lrow) * K + lcol;

  auto issue = [&](int buf, int k0) {
    __bf16* la = &As[buf][lrow * 32 + lcol];
    __bf16* lb = &Bs[buf][lrow * 32 + lcol];
    cp16(la,     ga + k0);
    cp16(la + 8, ga + k0 + 8);
    cp16(lb,     gw + k0);
    cp16(lb + 8, gw + k0 + 8);
  };

  f32x8 acc[4][2];
#pragma unroll
  for (int i = 0; i < 4; ++i)
#pragma unroll
    for (int j = 0; j < 2; ++j) acc[i][j] = zero8();

  issue(0, 0);
  for (int k0 = 0; k0 < K; k0 += 32) {
    const int buf = (k0 >> 5) & 1;
    wait_async0();          // this wave's LDS writes for buf are complete
    __syncthreads();        // everyone's writes visible; prev buf consumers done
    if (k0 + 32 < K) issue(buf ^ 1, k0 + 32);   // overlap with compute below

    bf16x16 bf0 = load_b_frag(&Bs[buf][(wn + 0)  * 32], 32);
    bf16x16 bf1 = load_b_frag(&Bs[buf][(wn + 16) * 32], 32);
#pragma unroll
    for (int i = 0; i < 4; ++i) {
      bf16x16 af = load_a_frag(&As[buf][(wm + i * 16) * 32], 32);
      acc[i][0] = wmma_bf16(af, bf0, acc[i][0]);
      acc[i][1] = wmma_bf16(af, bf1, acc[i][1]);
    }
  }

  const int cn = lane & 15;
  const int rb = (lane >> 4) * 8;
#pragma unroll
  for (int i = 0; i < 4; ++i) {
    const int row = m0 + wm + i * 16 + rb;
#pragma unroll
    for (int j = 0; j < 2; ++j) {
      const int col = n0 + wn + j * 16 + cn;
#pragma unroll
      for (int v = 0; v < 8; ++v)
        C[(size_t)(row + v) * N + col] = acc[i][j][v];
    }
  }
}

// ---------------------------------------------------------------------------
// RoPE + layout split. qkv fp32 [b,s,3,h,d] ->
//   qb, kb  bf16 (b,h,s,d) roped ; vt bf16 (b,h,d,s) ; kout fp32 (b,s,h,d)
// ---------------------------------------------------------------------------
__global__ __launch_bounds__(256) void rope_split(
    const float* __restrict__ qkv, __bf16* __restrict__ qb,
    __bf16* __restrict__ kb, __bf16* __restrict__ vt,
    float* __restrict__ kout)
{
  const unsigned int t = blockIdx.x * 256u + threadIdx.x;   // < B*S*NH*64
  const int i = t & 63;
  const int h = (t >> 6) & 15;
  const int s = (t >> 10) & 2047;
  const int b = t >> 21;

  const size_t base = ((size_t)(b * Ss + s)) * (3 * HID) + h * HD + 2 * i;
  const float q0 = qkv[base],            q1 = qkv[base + 1];
  const float k0 = qkv[base + HID],      k1 = qkv[base + HID + 1];
  const float v0 = qkv[base + 2 * HID],  v1 = qkv[base + 2 * HID + 1];

  const float inv = exp2f(-0.20762047f * (float)i);   // 10000^(-i/64)
  float sn, cs;
  __sincosf((float)s * inv, &sn, &cs);

  const float qr0 = q0 * cs - q1 * sn, qr1 = q0 * sn + q1 * cs;
  const float kr0 = k0 * cs - k1 * sn, kr1 = k0 * sn + k1 * cs;

  const size_t hs = (((size_t)(b * NH + h)) * Ss + s) * HD + 2 * i;
  qb[hs]     = (__bf16)qr0;
  qb[hs + 1] = (__bf16)qr1;
  kb[hs]     = (__bf16)kr0;
  kb[hs + 1] = (__bf16)kr1;

  const size_t ko = (((size_t)(b * Ss + s)) * NH + h) * HD + 2 * i;
  kout[ko]     = kr0;
  kout[ko + 1] = kr1;

  const size_t vo = (((size_t)(b * NH + h)) * HD + 2 * i) * Ss + s;
  vt[vo]      = (__bf16)v0;
  vt[vo + Ss] = (__bf16)v1;
}

// ---------------------------------------------------------------------------
// Flash attention: grid (S/128, B*NH), 256 threads. Double-buffered async K/V
// chunks, online softmax, WMMA for both GEMMs, bf16 output (b,s,h*HD+d).
// ---------------------------------------------------------------------------
__global__ __launch_bounds__(256) void flash_attn(
    const __bf16* __restrict__ Q, const __bf16* __restrict__ Kc,
    const __bf16* __restrict__ Vt, __bf16* __restrict__ O)
{
  __shared__ __bf16 Qs[128 * 128];        // [q][d]
  __shared__ __bf16 Ks[2][128 * 128];     // [key][d]
  __shared__ __bf16 Vs[2][128 * 128];     // [d][key]
  __shared__ float  Sc[128 * 128];
  __shared__ __bf16 Ps[128 * 128];
  __shared__ float  rowm[128], rowl[128], rowa[128];

  const int tid  = threadIdx.x;
  const int bh   = blockIdx.y;
  const int q0   = blockIdx.x * 128;
  const int lane = tid & 31;
  const int w    = tid >> 5;
  const int wm   = (w >> 2) * 64;
  const int wn   = (w & 3) * 32;
  const int cr   = tid >> 1;
  const int cc   = (tid & 1) * 64;
  const int cn   = lane & 15;
  const int rb   = (lane >> 4) * 8;

  const __bf16* gq = Q  + ((size_t)bh * Ss + q0 + cr) * HD + cc;
  const __bf16* gk = Kc + ((size_t)bh * Ss + cr) * HD + cc;
  const __bf16* gv = Vt + ((size_t)bh * HD + cr) * Ss + cc;

  auto issue_kv = [&](int buf, int kt) {
    __bf16* kd = &Ks[buf][cr * 128 + cc];
    __bf16* vd = &Vs[buf][cr * 128 + cc];
    const __bf16* ks = gk + (size_t)kt * 128 * HD;
    const __bf16* vs = gv + kt * 128;
#pragma unroll
    for (int u = 0; u < 8; ++u) {
      cp16(kd + u * 8, ks + u * 8);
      cp16(vd + u * 8, vs + u * 8);
    }
  };

  // prologue: Q tile + chunk 0
  {
    __bf16* qd = &Qs[cr * 128 + cc];
#pragma unroll
    for (int u = 0; u < 8; ++u) cp16(qd + u * 8, gq + u * 8);
  }
  issue_kv(0, 0);
  if (tid < 128) { rowm[tid] = -3.0e38f; rowl[tid] = 0.0f; }

  f32x8 accO[4][2];
#pragma unroll
  for (int i = 0; i < 4; ++i)
#pragma unroll
    for (int j = 0; j < 2; ++j) accO[i][j] = zero8();

  for (int kt = 0; kt < Ss / 128; ++kt) {
    const int buf = kt & 1;
    wait_async0();
    __syncthreads();
    if (kt + 1 < Ss / 128) issue_kv(buf ^ 1, kt + 1);   // overlap next chunk

    // ----- scores = Q @ K^T ------------------------------------------------
    f32x8 accS[4][2];
#pragma unroll
    for (int i = 0; i < 4; ++i)
#pragma unroll
      for (int j = 0; j < 2; ++j) accS[i][j] = zero8();

#pragma unroll
    for (int k0 = 0; k0 < HD; k0 += 32) {
      bf16x16 bf0 = load_b_frag(&Ks[buf][(wn + 0)  * 128 + k0], 128);
      bf16x16 bf1 = load_b_frag(&Ks[buf][(wn + 16) * 128 + k0], 128);
#pragma unroll
      for (int i = 0; i < 4; ++i) {
        bf16x16 af = load_a_frag(&Qs[(wm + i * 16) * 128 + k0], 128);
        accS[i][0] = wmma_bf16(af, bf0, accS[i][0]);
        accS[i][1] = wmma_bf16(af, bf1, accS[i][1]);
      }
    }
#pragma unroll
    for (int i = 0; i < 4; ++i) {
      const int rr = wm + i * 16 + rb;
#pragma unroll
      for (int j = 0; j < 2; ++j) {
        const int col = wn + j * 16 + cn;
#pragma unroll
        for (int v = 0; v < 8; ++v)
          Sc[(rr + v) * 128 + col] = accS[i][j][v] * SCALE;
      }
    }
    __syncthreads();

    // ----- online softmax per row -----------------------------------------
    if (tid < 128) {
      const float* sr = Sc + tid * 128;
      float m_old = rowm[tid], mx = m_old;
#pragma unroll 8
      for (int c = 0; c < 128; ++c) mx = fmaxf(mx, sr[c]);
      const float alpha = __expf(m_old - mx);
      float sum = 0.0f;
      __bf16* pr = Ps + tid * 128;
#pragma unroll 8
      for (int c = 0; c < 128; ++c) {
        const float p = __expf(sr[c] - mx);
        sum += p;
        pr[c] = (__bf16)p;
      }
      rowl[tid] = rowl[tid] * alpha + sum;
      rowm[tid] = mx;
      rowa[tid] = alpha;
    }
    __syncthreads();

    // ----- rescale O, then O += P @ V -------------------------------------
#pragma unroll
    for (int i = 0; i < 4; ++i) {
      const int rr = wm + i * 16 + rb;
      float a[8];
#pragma unroll
      for (int v = 0; v < 8; ++v) a[v] = rowa[rr + v];
#pragma unroll
      for (int j = 0; j < 2; ++j)
#pragma unroll
        for (int v = 0; v < 8; ++v) accO[i][j][v] *= a[v];
    }
#pragma unroll
    for (int k0 = 0; k0 < 128; k0 += 32) {
      bf16x16 bf0 = load_b_frag(&Vs[buf][(wn + 0)  * 128 + k0], 128);
      bf16x16 bf1 = load_b_frag(&Vs[buf][(wn + 16) * 128 + k0], 128);
#pragma unroll
      for (int i = 0; i < 4; ++i) {
        bf16x16 af = load_a_frag(&Ps[(wm + i * 16) * 128 + k0], 128);
        accO[i][0] = wmma_bf16(af, bf0, accO[i][0]);
        accO[i][1] = wmma_bf16(af, bf1, accO[i][1]);
      }
    }
  }
  __syncthreads();

  // ----- epilogue: O /= l, bf16, layout (b, s, h*HD + d) -------------------
  const int b = bh >> 4, h = bh & 15;
#pragma unroll
  for (int i = 0; i < 4; ++i) {
    const int rr = wm + i * 16 + rb;
    float inv[8];
#pragma unroll
    for (int v = 0; v < 8; ++v) inv[v] = 1.0f / rowl[rr + v];
#pragma unroll
    for (int j = 0; j < 2; ++j) {
      const int col = wn + j * 16 + cn;
#pragma unroll
      for (int v = 0; v < 8; ++v) {
        const size_t idx = ((size_t)(b * Ss + q0 + rr + v)) * HID + h * HD + col;
        O[idx] = (__bf16)(accO[i][j][v] * inv[v]);
      }
    }
  }
}

// ---------------------------------------------------------------------------
extern "C" void kernel_launch(void* const* d_in, const int* in_sizes, int n_in,
                              void* d_out, int out_size, void* d_ws, size_t ws_size,
                              hipStream_t stream) {
  const float* x     = (const float*)d_in[0];   // (B,S,HID)
  const float* w_qkv = (const float*)d_in[1];   // (3*HID, HID)
  const float* w_out = (const float*)d_in[2];   // (HID, HID)

  float* out  = (float*)d_out;                  // (B,S,HID)
  float* kout = out + (size_t)MTOK * HID;       // (B,S,NH,HD)

  // workspace partition
  float*  qkv   = (float*)d_ws;                               // MTOK x 3*HID fp32
  __bf16* xb    = (__bf16*)(qkv + (size_t)MTOK * 3 * HID);    // MTOK x HID
  __bf16* wqkvb = xb    + (size_t)MTOK * HID;                 // 3*HID x HID
  __bf16* woutb = wqkvb + (size_t)3 * HID * HID;              // HID x HID
  __bf16* qb    = woutb + (size_t)HID * HID;                  // (b,h,s,d)
  __bf16* kb    = qb    + (size_t)Bb * NH * Ss * HD;
  __bf16* vt    = kb    + (size_t)Bb * NH * Ss * HD;          // (b,h,d,s)
  __bf16* attnb = vt    + (size_t)Bb * NH * Ss * HD;          // MTOK x HID

  // 0) convert operands to bf16 (RNE) once
  f32_to_bf16<<<(MTOK * HID / 4 + 255) / 256, 256, 0, stream>>>(x, xb, MTOK * HID / 4);
  f32_to_bf16<<<(3 * HID * HID / 4 + 255) / 256, 256, 0, stream>>>(w_qkv, wqkvb, 3 * HID * HID / 4);
  f32_to_bf16<<<(HID * HID / 4 + 255) / 256, 256, 0, stream>>>(w_out, woutb, HID * HID / 4);

  // 1) qkv = x @ w_qkv^T
  gemm_bf16nt<<<dim3(3 * HID / 128, MTOK / 128), 256, 0, stream>>>(
      xb, wqkvb, qkv, MTOK, 3 * HID, HID);

  // 2) RoPE + split/transpose (+ fp32 roped k to output slot 2)
  rope_split<<<(Bb * Ss * NH * 64) / 256, 256, 0, stream>>>(qkv, qb, kb, vt, kout);

  // 3) flash attention (bf16 out)
  flash_attn<<<dim3(Ss / 128, Bb * NH), 256, 0, stream>>>(qb, kb, vt, attnb);

  // 4) out = attn @ w_out^T
  gemm_bf16nt<<<dim3(HID / 128, MTOK / 128), 256, 0, stream>>>(
      attnb, woutb, out, MTOK, HID, HID);
}
